// CrossAttentionBlock_77257871721114
// MI455X (gfx1250) — compile-verified, matrix operand order checked by reading, not run on hardware
//
#include <hip/hip_runtime.h>
#include <hip/hip_bf16.h>

// ---------------------------------------------------------------------------
// CrossAttentionBlock on MI455X (gfx1250), wave32 + v_wmma_f32_16x16x32_f16.
// Addressing uses uniform (SGPR) bases + one 32-bit lane voffset + instruction
// immediates so fragment loads are saddr-form b128s (minimal VGPR pressure).
// ---------------------------------------------------------------------------

typedef _Float16 v8h  __attribute__((ext_vector_type(8)));
typedef _Float16 v16h __attribute__((ext_vector_type(16)));
typedef float    v8f  __attribute__((ext_vector_type(8)));

#define CH   512
#define CTX  768
#define PIX  4096
#define NH   8
#define DH   64

__device__ __forceinline__ v8f zero8() {
    v8f z = {0.f,0.f,0.f,0.f,0.f,0.f,0.f,0.f};
    return z;
}

// Build a 16x32 f16 A/B fragment for one lane from two contiguous 8-half chunks.
// Per ISA 7.12.2: lanes 0-15 hold K = k0..k0+7 (lo) and k0+16..k0+23 (hi);
// lanes 16-31 hold K = k0+8.. and k0+24.. -> caller bakes the +8 into offsets.
__device__ __forceinline__ v16h ldfrag(const _Float16* lo, const _Float16* hi) {
    v8h a = *(const v8h*)lo;
    v8h b = *(const v8h*)hi;
    v16h r;
#pragma unroll
    for (int e = 0; e < 8; ++e) { r[e] = a[e]; r[e + 8] = b[e]; }
    return r;
}

// ---------------------------------------------------------------------------
// Elementwise f32 -> f16 convert (weights).
// ---------------------------------------------------------------------------
__global__ void cvt16_kernel(const float* __restrict__ s, _Float16* __restrict__ d, int n) {
    int i = blockIdx.x * 256 + threadIdx.x;
    if (i < n) d[i] = (_Float16)s[i];
}

// ---------------------------------------------------------------------------
// GroupNorm pass 1: per-group mean / rstd.  One block per group.
// ---------------------------------------------------------------------------
__global__ void gn_stats_kernel(const float* __restrict__ x, float* __restrict__ stats) {
    __shared__ float ssum[256], ssq[256];
    int g = blockIdx.x, tid = threadIdx.x;
    const float* p = x + (size_t)g * 65536;
    float s = 0.f, q = 0.f;
    for (int i = tid; i < 65536; i += 256) { float v = p[i]; s += v; q += v * v; }
    ssum[tid] = s; ssq[tid] = q;
    __syncthreads();
    for (int st = 128; st > 0; st >>= 1) {
        if (tid < st) { ssum[tid] += ssum[tid + st]; ssq[tid] += ssq[tid + st]; }
        __syncthreads();
    }
    if (tid == 0) {
        float mu  = ssum[0] * (1.f / 65536.f);
        float var = ssq[0] * (1.f / 65536.f) - mu * mu;
        stats[g * 2]     = mu;
        stats[g * 2 + 1] = rsqrtf(var + 1e-5f);
    }
}

// GroupNorm pass 2: normalize + affine, write transposed f16 hT[p, c].
__global__ void gn_apply_t_kernel(const float* __restrict__ x, const float* __restrict__ stats,
                                  const float* __restrict__ w, const float* __restrict__ b,
                                  _Float16* __restrict__ hT) {
    int idx = blockIdx.x * 256 + threadIdx.x;   // over CH*PIX, channel-major
    int c = idx >> 12;
    int p = idx & 4095;
    int g = c >> 4;
    float mu = stats[g * 2], rs = stats[g * 2 + 1];
    float v = (x[idx] - mu) * rs * w[c] + b[c];
    hT[(size_t)p * CH + c] = (_Float16)v;
}

// Transpose + convert: src f32 [C, 4096] -> dst f16 [4096, C].
__global__ void t_cvt_kernel(const float* __restrict__ src, _Float16* __restrict__ dst, int C) {
    int idx = blockIdx.x * 256 + threadIdx.x;
    int c = idx >> 12;
    int p = idx & 4095;
    dst[(size_t)p * C + c] = (_Float16)src[idx];
}

// ---------------------------------------------------------------------------
// WMMA GEMM, 32x64 output per wave (2x4 tiles of 16x16).
//   A: [M x K] f16 row-major (weights), B: [N x K] f16 row-major (act^T)
// Uniform tile base pointers (SGPRs) + one lane voffset VGPR.
// MODE 0: out16[n*M + m]  (transposed f16, b128 stores)  -> Qt/Kt
// MODE 1: out16[m*N + n]  (normal f16)                   -> V
// MODE 2: out32[m*N + n] = acc + bias + resid            -> d_out
// ---------------------------------------------------------------------------
template <int MODE>
__global__ __launch_bounds__(256)
void gemm_wmma_kernel(const _Float16* __restrict__ A, const _Float16* __restrict__ B,
                      const float* __restrict__ bias,
                      int M, int N, int K,
                      _Float16* __restrict__ out16,
                      float* __restrict__ out32,
                      const float* __restrict__ resid) {
    int wave = (blockIdx.x * blockDim.x + threadIdx.x) >> 5;
    int lane = threadIdx.x & 31;
    int ngrp = N >> 6;                      // 64-wide column groups
    int mg   = wave / ngrp;
    int ng   = wave - mg * ngrp;
    int m0 = mg << 5, n0 = ng << 6;         // 32 rows x 64 cols per wave
    int half = lane >> 4;
    int idx  = lane & 15;

    // Wave-uniform tile bases (SGPRs); single lane-variant 32-bit offset.
    const _Float16* A0 = A + (size_t)m0 * K;
    const _Float16* A1 = A0 + (size_t)16 * K;
    const _Float16* B0 = B + (size_t)n0 * K;
    const _Float16* B1 = B0 + (size_t)16 * K;
    const _Float16* B2 = B1 + (size_t)16 * K;
    const _Float16* B3 = B2 + (size_t)16 * K;
    unsigned aoff = (unsigned)idx * (unsigned)K + half * 8;

    v8f acc[2][4];
#pragma unroll
    for (int tm = 0; tm < 2; ++tm)
#pragma unroll
        for (int tn = 0; tn < 4; ++tn) acc[tm][tn] = zero8();

    for (int k0 = 0; k0 < K; k0 += 32) {
        unsigned o = aoff + (unsigned)k0;
        v16h a0 = ldfrag(A0 + o, A0 + o + 16);
        v16h a1 = ldfrag(A1 + o, A1 + o + 16);
        v16h b0 = ldfrag(B0 + o, B0 + o + 16);
        v16h b1 = ldfrag(B1 + o, B1 + o + 16);
        v16h b2 = ldfrag(B2 + o, B2 + o + 16);
        v16h b3 = ldfrag(B3 + o, B3 + o + 16);
        acc[0][0] = __builtin_amdgcn_wmma_f32_16x16x32_f16(false, a0, false, b0, (short)0, acc[0][0], false, false);
        acc[0][1] = __builtin_amdgcn_wmma_f32_16x16x32_f16(false, a0, false, b1, (short)0, acc[0][1], false, false);
        acc[0][2] = __builtin_amdgcn_wmma_f32_16x16x32_f16(false, a0, false, b2, (short)0, acc[0][2], false, false);
        acc[0][3] = __builtin_amdgcn_wmma_f32_16x16x32_f16(false, a0, false, b3, (short)0, acc[0][3], false, false);
        acc[1][0] = __builtin_amdgcn_wmma_f32_16x16x32_f16(false, a1, false, b0, (short)0, acc[1][0], false, false);
        acc[1][1] = __builtin_amdgcn_wmma_f32_16x16x32_f16(false, a1, false, b1, (short)0, acc[1][1], false, false);
        acc[1][2] = __builtin_amdgcn_wmma_f32_16x16x32_f16(false, a1, false, b2, (short)0, acc[1][2], false, false);
        acc[1][3] = __builtin_amdgcn_wmma_f32_16x16x32_f16(false, a1, false, b3, (short)0, acc[1][3], false, false);
    }

#pragma unroll
    for (int tm = 0; tm < 2; ++tm) {
        int rbase = m0 + tm * 16 + half * 8;   // 8 consecutive output rows
        float bv[8];
#pragma unroll
        for (int r = 0; r < 8; ++r) bv[r] = bias[rbase + r];
#pragma unroll
        for (int tn = 0; tn < 4; ++tn) {
            int col = n0 + tn * 16 + idx;
            if (MODE == 0) {
                v8h t;
#pragma unroll
                for (int r = 0; r < 8; ++r) t[r] = (_Float16)(acc[tm][tn][r] + bv[r]);
                *(v8h*)(out16 + (size_t)col * M + rbase) = t;
            } else if (MODE == 1) {
#pragma unroll
                for (int r = 0; r < 8; ++r)
                    out16[(size_t)(rbase + r) * N + col] = (_Float16)(acc[tm][tn][r] + bv[r]);
            } else {
#pragma unroll
                for (int r = 0; r < 8; ++r) {
                    size_t o = (size_t)(rbase + r) * N + col;
                    out32[o] = acc[tm][tn][r] + bv[r] + resid[o];
                }
            }
        }
    }
}

// ---------------------------------------------------------------------------
// Online-softmax update for one 16-query tile (two 16-key Sᵀ sub-fragments).
// Scores arrive raw and are scaled into the log2 domain here.  Returns the rescale factor
// for the output accumulator; fills bp = P as the B-fragment of the P·V WMMA.
// ---------------------------------------------------------------------------
__device__ __forceinline__ float softmax_update(v8f s0, v8f s1, float& mrow, float& lrow,
                                                v16h& bp) {
    const float sc2 = 0.18033688011112042f;   // (1/sqrt(64)) * log2(e)
    float mx = -3.0e38f;
#pragma unroll
    for (int r = 0; r < 8; ++r) {
        s0[r] *= sc2; s1[r] *= sc2;
        mx = fmaxf(mx, fmaxf(s0[r], s1[r]));
    }
    mx = fmaxf(mx, __shfl_xor(mx, 16, 32));
    float mnew = fmaxf(mrow, mx);
    float corr = __builtin_amdgcn_exp2f(mrow - mnew);
    float ls = 0.f;
#pragma unroll
    for (int r = 0; r < 8; ++r) {
        float p0 = __builtin_amdgcn_exp2f(s0[r] - mnew);
        float p1 = __builtin_amdgcn_exp2f(s1[r] - mnew);
        ls += p0 + p1;
        bp[r]     = (_Float16)p0;   // K(j) = half*8 + r      (subtile 0)
        bp[r + 8] = (_Float16)p1;   // K(j) = 16 + half*8 + r (subtile 1)
    }
    ls += __shfl_xor(ls, 16, 32);
    lrow = lrow * corr + ls;
    mrow = mnew;
    return corr;
}

// ---------------------------------------------------------------------------
// Flash cross-attention: one wave per (head, 32-query group).  Two query
// tiles (A/B) share every K and V fragment; softmaxes run sequentially so
// only one tile's S fragments are ever live.  Uniform head-sliced bases +
// two 32-bit voffsets; K-subtile / d-chunk deltas are instruction immediates.
//   Qt, Kt : [4096, 512] f16   V : [512, 4096] f16   AO : [4096, 512] f16
// ---------------------------------------------------------------------------
__global__ __launch_bounds__(256)
void attn_wmma_kernel(const _Float16* __restrict__ Qt,
                      const _Float16* __restrict__ Kt,
                      const _Float16* __restrict__ V,
                      _Float16* __restrict__ AO) {
    int wave = (blockIdx.x * blockDim.x + threadIdx.x) >> 5;
    int lane = threadIdx.x & 31;
    int head = wave >> 7;              // 128 query groups per head
    int grp  = wave & 127;
    int i0   = grp << 5;               // 32 queries per wave
    int half = lane >> 4;
    int idx  = lane & 15;

    // Uniform head-sliced bases (SGPRs).
    const _Float16* QtH = Qt + head * DH;
    const _Float16* KtH = Kt + head * DH;
    const _Float16* VH  = V + (size_t)(head * DH) * PIX;
    _Float16*       AOH = AO + head * DH;

    // Q B-fragments for both query tiles (fixed for the whole wave).
    unsigned qo = (unsigned)(i0 + idx) * CH + half * 8;
    v16h bqA0 = ldfrag(QtH + qo,      QtH + qo + 16);
    v16h bqA1 = ldfrag(QtH + qo + 32, QtH + qo + 48);
    v16h bqB0 = ldfrag(QtH + qo + 16 * CH,      QtH + qo + 16 * CH + 16);
    v16h bqB1 = ldfrag(QtH + qo + 16 * CH + 32, QtH + qo + 16 * CH + 48);

    v8f oA[4], oB[4];
#pragma unroll
    for (int dc = 0; dc < 4; ++dc) { oA[dc] = zero8(); oB[dc] = zero8(); }
    float mA = -3.0e38f, lA = 0.f, mB = -3.0e38f, lB = 0.f;

    unsigned ko = (unsigned)idx * CH + half * 8;   // += 32*CH per j-tile
    unsigned vo = (unsigned)idx * PIX + half * 8;  // += 32 per j-tile

    for (int j0 = 0; j0 < PIX; j0 += 32) {
        // ---- K fragments (shared by both query tiles); subtile delta = imm.
        v16h k00 = ldfrag(KtH + ko,      KtH + ko + 16);
        v16h k01 = ldfrag(KtH + ko + 32, KtH + ko + 48);
        v16h k10 = ldfrag(KtH + ko + 16 * CH,      KtH + ko + 16 * CH + 16);
        v16h k11 = ldfrag(KtH + ko + 16 * CH + 32, KtH + ko + 16 * CH + 48);

        // ---- tile A: Sᵀ (4 WMMA) then softmax (frees sA before tile B)
        v16h bpA, bpB;
        {
            v8f sA0 = zero8(), sA1 = zero8();
            sA0 = __builtin_amdgcn_wmma_f32_16x16x32_f16(false, k00, false, bqA0, (short)0, sA0, false, false);
            sA1 = __builtin_amdgcn_wmma_f32_16x16x32_f16(false, k10, false, bqA0, (short)0, sA1, false, false);
            sA0 = __builtin_amdgcn_wmma_f32_16x16x32_f16(false, k01, false, bqA1, (short)0, sA0, false, false);
            sA1 = __builtin_amdgcn_wmma_f32_16x16x32_f16(false, k11, false, bqA1, (short)0, sA1, false, false);
            float corrA = softmax_update(sA0, sA1, mA, lA, bpA);
#pragma unroll
            for (int dc = 0; dc < 4; ++dc)
#pragma unroll
                for (int r = 0; r < 8; ++r) oA[dc][r] *= corrA;
        }
        // ---- tile B
        {
            v8f sB0 = zero8(), sB1 = zero8();
            sB0 = __builtin_amdgcn_wmma_f32_16x16x32_f16(false, k00, false, bqB0, (short)0, sB0, false, false);
            sB1 = __builtin_amdgcn_wmma_f32_16x16x32_f16(false, k10, false, bqB0, (short)0, sB1, false, false);
            sB0 = __builtin_amdgcn_wmma_f32_16x16x32_f16(false, k01, false, bqB1, (short)0, sB0, false, false);
            sB1 = __builtin_amdgcn_wmma_f32_16x16x32_f16(false, k11, false, bqB1, (short)0, sB1, false, false);
            float corrB = softmax_update(sB0, sB1, mB, lB, bpB);
#pragma unroll
            for (int dc = 0; dc < 4; ++dc)
#pragma unroll
                for (int r = 0; r < 8; ++r) oB[dc][r] *= corrB;
        }

        // ---- Oᵀ += V·Pᵀ : 4 d-chunks x 2 query tiles; d-chunk delta = imm.
#pragma unroll
        for (int dc = 0; dc < 4; ++dc) {
            v16h av = ldfrag(VH + vo + dc * 16 * PIX, VH + vo + dc * 16 * PIX + 16);
            oA[dc] = __builtin_amdgcn_wmma_f32_16x16x32_f16(false, av, false, bpA, (short)0, oA[dc], false, false);
            oB[dc] = __builtin_amdgcn_wmma_f32_16x16x32_f16(false, av, false, bpB, (short)0, oB[dc], false, false);
        }

        ko += 32 * CH;
        vo += 32;
    }

    // ---- finalize: divide by row sums, contiguous b128 stores to AO[p, c].
    float invA = 1.f / lA, invB = 1.f / lB;
    unsigned ao = (unsigned)(i0 + idx) * CH + half * 8;
#pragma unroll
    for (int dc = 0; dc < 4; ++dc) {
        v8h tA, tB;
#pragma unroll
        for (int r = 0; r < 8; ++r) {
            tA[r] = (_Float16)(oA[dc][r] * invA);
            tB[r] = (_Float16)(oB[dc][r] * invB);
        }
        *(v8h*)(AOH + ao + dc * 16) = tA;
        *(v8h*)(AOH + ao + 16 * CH + dc * 16) = tB;
    }
}

// ---------------------------------------------------------------------------
// Host launcher
// ---------------------------------------------------------------------------
extern "C" void kernel_launch(void* const* d_in, const int* in_sizes, int n_in,
                              void* d_out, int out_size, void* d_ws, size_t ws_size,
                              hipStream_t stream) {
    (void)in_sizes; (void)n_in; (void)out_size; (void)ws_size;
    const float* x    = (const float*)d_in[0];
    const float* ctx  = (const float*)d_in[1];
    const float* gnw  = (const float*)d_in[2];
    const float* gnb  = (const float*)d_in[3];
    const float* wq   = (const float*)d_in[4];
    const float* bq   = (const float*)d_in[5];
    const float* wk   = (const float*)d_in[6];
    const float* bk   = (const float*)d_in[7];
    const float* wv   = (const float*)d_in[8];
    const float* bv   = (const float*)d_in[9];
    const float* wo   = (const float*)d_in[10];
    const float* bo   = (const float*)d_in[11];
    float* out = (float*)d_out;

    char* ws = (char*)d_ws;
    size_t off = 0;
    auto carve = [&](size_t bytes) -> void* {
        void* p = ws + off;
        off += (bytes + 255) & ~(size_t)255;
        return p;
    };
    float*     stats = (float*)    carve(32 * 2 * sizeof(float));
    _Float16*  hT    = (_Float16*) carve((size_t)PIX * CH  * 2);   // [4096,512]
    _Float16*  ctxT  = (_Float16*) carve((size_t)PIX * CTX * 2);   // [4096,768]
    _Float16*  wq16  = (_Float16*) carve((size_t)CH * CH  * 2);
    _Float16*  wk16  = (_Float16*) carve((size_t)CH * CTX * 2);
    _Float16*  wv16  = (_Float16*) carve((size_t)CH * CTX * 2);
    _Float16*  wo16  = (_Float16*) carve((size_t)CH * CH  * 2);
    _Float16*  Qt    = (_Float16*) carve((size_t)PIX * CH * 2);    // [4096,512]
    _Float16*  Kt    = (_Float16*) carve((size_t)PIX * CH * 2);    // [4096,512]
    _Float16*  Vn    = (_Float16*) carve((size_t)CH * PIX * 2);    // [512,4096]
    _Float16*  AO    = (_Float16*) carve((size_t)PIX * CH * 2);    // [4096,512]

    // 1) weight conversions
    cvt16_kernel<<<(CH*CH  + 255)/256, 256, 0, stream>>>(wq, wq16, CH*CH);
    cvt16_kernel<<<(CH*CTX + 255)/256, 256, 0, stream>>>(wk, wk16, CH*CTX);
    cvt16_kernel<<<(CH*CTX + 255)/256, 256, 0, stream>>>(wv, wv16, CH*CTX);
    cvt16_kernel<<<(CH*CH  + 255)/256, 256, 0, stream>>>(wo, wo16, CH*CH);

    // 2) groupnorm -> hT (transposed f16), ctx -> ctxT
    gn_stats_kernel  <<<32, 256, 0, stream>>>(x, stats);
    gn_apply_t_kernel<<<CH  * PIX / 256, 256, 0, stream>>>(x, stats, gnw, gnb, hT);
    t_cvt_kernel     <<<CTX * PIX / 256, 256, 0, stream>>>(ctx, ctxT, CTX);

    // 3) projections: 32x64 tile/wave -> (512/32)*(4096/64)=1024 waves, 128 blocks
    int blocks = (CH / 32) * (PIX / 64) / 8;
    gemm_wmma_kernel<0><<<blocks, 256, 0, stream>>>(wq16, hT,   bq, CH, PIX, CH,  Qt, nullptr, nullptr);
    gemm_wmma_kernel<0><<<blocks, 256, 0, stream>>>(wk16, ctxT, bk, CH, PIX, CTX, Kt, nullptr, nullptr);
    gemm_wmma_kernel<1><<<blocks, 256, 0, stream>>>(wv16, ctxT, bv, CH, PIX, CTX, Vn, nullptr, nullptr);

    // 4) flash attention: 8 heads x 128 query groups = 1024 waves -> 128 blocks
    attn_wmma_kernel<<<128, 256, 0, stream>>>(Qt, Kt, Vn, AO);

    // 5) output projection + bias + residual -> d_out (f32)
    gemm_wmma_kernel<2><<<blocks, 256, 0, stream>>>(wo16, AO, bo, CH, PIX, CH, nullptr, out, x);
}
